// CombNet_v1_48258252538509
// MI455X (gfx1250) — matrix operand order repaired and checked by reading, not compiled, not run on hardware
//
#include <hip/hip_runtime.h>
#include <hip/hip_fp16.h>
#include <hip/hip_bf16.h>

// ---------------------------------------------------------------------------
// CDNA5 (gfx1250) Graph U-Net. Heavy matmuls -> wave32 WMMA GEMM using
// v_wmma_f32_16x16x32_f16, 32x32 register tile per wave, all fragment loads
// as contiguous b128 (B operand supplied column-major; the dominant GEMM's
// B is the symmetric adjacency so it is its own transpose).
// Adjacency stored f16 (exact: 0/1 and small integer co-neighbor counts),
// L2-resident on MI455X (4096^2 f16 = 32MB << 192MB L2).
// ---------------------------------------------------------------------------

typedef _Float16 half_t;
typedef __attribute__((ext_vector_type(16))) _Float16 v16h;
typedef __attribute__((ext_vector_type(8)))  _Float16 v8h;
typedef __attribute__((ext_vector_type(8)))  float    v8f;

#define N0    4096
#define NEDGE 65536
#define INC   32
#define HIDC  64
#define OUTC  32

// ---- fragment loaders: contiguous 16B vector loads ------------------------
// A-matrix 16x32 f16 layout (ISA 7.12.2): lane L (0-15): row=L, K 0-7 & 16-23;
// lane 16-31: row=L-16... here per-lane kb = k0 + ((lane>>4)<<3).
__device__ __forceinline__ v16h load_a_frag(const half_t* __restrict__ A,
                                            int row, int K, int kb) {
  const v8h* p = (const v8h*)(A + (size_t)row * K + kb);
  v8h lo = p[0];        // K offsets +0..7
  v8h hi = p[2];        // K offsets +16..23
  return __builtin_shufflevector(lo, hi, 0, 1, 2, 3, 4, 5, 6, 7,
                                 8, 9, 10, 11, 12, 13, 14, 15);
}
// B-matrix 32x16 f16 layout: lane L: col=L&15, K base (L>>4)*8, K offsets
// +0..7 & +16..23 (matches A) when B is stored column-major (Bt[N x K]).
__device__ __forceinline__ v16h load_bt_frag(const half_t* __restrict__ Bt,
                                             int col, int K, int kb) {
  const v8h* p = (const v8h*)(Bt + (size_t)col * K + kb);
  return __builtin_shufflevector(p[0], p[2], 0, 1, 2, 3, 4, 5, 6, 7,
                                 8, 9, 10, 11, 12, 13, 14, 15);
}

__device__ __forceinline__ void store_tile_f32(float* __restrict__ Cf,
                                               const v8f& acc, int Nn,
                                               int rbase, int col) {
#pragma unroll
  for (int r = 0; r < 8; ++r)
    Cf[(size_t)(rbase + r) * Nn + col] = acc[r];
}
__device__ __forceinline__ void store_tile_f16_diag0(half_t* __restrict__ Ch,
                                                     const v8f& acc, int Nn,
                                                     int rbase, int col) {
#pragma unroll
  for (int r = 0; r < 8; ++r) {
    int row = rbase + r;
    Ch[(size_t)row * Nn + col] = (row == col) ? (half_t)0.0f : (half_t)acc[r];
  }
}

// ---------------------------------------------------------------------------
// WMMA GEMM: C[M x Nn] = A[M x K] @ B[K x Nn], with B passed COLUMN-MAJOR as
// Bt[Nn x K]. One wave computes a 32x32 C tile (4 accumulators, 4 WMMAs per
// 32-deep K step). MODE 0: f32 store. MODE 1: f16 store w/ zero diagonal.
// ---------------------------------------------------------------------------
template <int MODE>
__global__ void k_gemm_f16(const half_t* __restrict__ A,
                           const half_t* __restrict__ Bt,
                           float* __restrict__ Cf,
                           half_t* __restrict__ Ch,
                           int M, int Nn, int K) {
  int wid    = (int)((blockIdx.x * blockDim.x + threadIdx.x) >> 5);
  int lane   = threadIdx.x & 31;
  int tilesN = Nn >> 5;
  int tiles  = (M >> 5) * tilesN;
  if (wid >= tiles) return;               // uniform per wave: EXEC all-1s
  int tm = (wid / tilesN) << 5;
  int tn = (wid % tilesN) << 5;

  int arow0 = tm + (lane & 15);
  int arow1 = arow0 + 16;
  int akb   = (lane >> 4) << 3;           // 0 / 8
  int bcol0 = tn + (lane & 15);
  int bcol1 = bcol0 + 16;
  int bkb   = (lane >> 4) << 3;           // 0 / 8

  v8f c00 = {}, c01 = {}, c10 = {}, c11 = {};
  for (int k0 = 0; k0 < K; k0 += 32) {
    v16h a0 = load_a_frag(A, arow0, K, k0 + akb);
    v16h a1 = load_a_frag(A, arow1, K, k0 + akb);
    v16h b0 = load_bt_frag(Bt, bcol0, K, k0 + bkb);
    v16h b1 = load_bt_frag(Bt, bcol1, K, k0 + bkb);
    c00 = __builtin_amdgcn_wmma_f32_16x16x32_f16(false, a0, false, b0,
                                                 (short)0, c00, false, false);
    c01 = __builtin_amdgcn_wmma_f32_16x16x32_f16(false, a0, false, b1,
                                                 (short)0, c01, false, false);
    c10 = __builtin_amdgcn_wmma_f32_16x16x32_f16(false, a1, false, b0,
                                                 (short)0, c10, false, false);
    c11 = __builtin_amdgcn_wmma_f32_16x16x32_f16(false, a1, false, b1,
                                                 (short)0, c11, false, false);
  }

  int rb0 = tm + ((lane >> 4) << 3);      // C tile: vgpr r -> row rb + r
  int rb1 = rb0 + 16;
  int cl0 = tn + (lane & 15);
  int cl1 = cl0 + 16;
  if (MODE == 0) {
    store_tile_f32(Cf, c00, Nn, rb0, cl0);
    store_tile_f32(Cf, c01, Nn, rb0, cl1);
    store_tile_f32(Cf, c10, Nn, rb1, cl0);
    store_tile_f32(Cf, c11, Nn, rb1, cl1);
  } else {
    store_tile_f16_diag0(Ch, c00, Nn, rb0, cl0);
    store_tile_f16_diag0(Ch, c01, Nn, rb0, cl1);
    store_tile_f16_diag0(Ch, c10, Nn, rb1, cl0);
    store_tile_f16_diag0(Ch, c11, Nn, rb1, cl1);
  }
}

// ---------------------------------------------------------------------------
// Elementwise / graph utility kernels
// ---------------------------------------------------------------------------
__global__ void k_fill_h(half_t* p, int n) {
  int i = blockIdx.x * blockDim.x + threadIdx.x;
  if (i < n) p[i] = (half_t)0.0f;
}

__global__ void k_build_adj(const int* __restrict__ ei, half_t* __restrict__ A,
                            int E, int n) {
  int e = blockIdx.x * blockDim.x + threadIdx.x;
  if (e >= E) return;
  int s = ei[e], d = ei[E + e];
  if (s != d) {                      // (1-eye) masks self loops
    A[(size_t)s * n + d] = (half_t)1.0f;
    A[(size_t)d * n + s] = (half_t)1.0f;
  }
}

__global__ void k_rowsum_dinv(const half_t* __restrict__ A,
                              float* __restrict__ dinv, int n) {
  int i = blockIdx.x * blockDim.x + threadIdx.x;
  if (i >= n) return;
  const v8h* row = (const v8h*)(A + (size_t)i * n);
  float s = 2.0f;                    // improved=True self-loop weight 2
  for (int j = 0; j < (n >> 3); ++j) {
    v8h v = row[j];
#pragma unroll
    for (int t = 0; t < 8; ++t) s += (float)v[t];
  }
  dinv[i] = (s > 0.0f) ? rsqrtf(s) : 0.0f;
}

__global__ void k_f32_to_f16(const float* __restrict__ x,
                             half_t* __restrict__ y, int n) {
  int i = blockIdx.x * blockDim.x + threadIdx.x;
  if (i < n) y[i] = (half_t)x[i];
}

// W[rows x cols] f32 -> Wt[cols x rows] f16 (column-major B operand)
__global__ void k_f32_to_f16_T(const float* __restrict__ x,
                               half_t* __restrict__ y, int rows, int cols) {
  int idx = blockIdx.x * blockDim.x + threadIdx.x;
  if (idx >= rows * cols) return;
  int r = idx / cols, c = idx % cols;
  y[(size_t)c * rows + r] = (half_t)x[idx];
}

// ys_t[c x n] = (dinv[i] * xw[i,c])^T  (column-major B operand for A @ ys)
__global__ void k_scale_rows_f16_T(const float* __restrict__ xw,
                                   const float* __restrict__ dinv,
                                   half_t* __restrict__ yt, int n, int c) {
  int idx = blockIdx.x * blockDim.x + threadIdx.x;
  if (idx >= n * c) return;
  int i = idx / c, j = idx % c;
  yt[(size_t)j * n + i] = (half_t)(dinv[i] * xw[idx]);
}

// out = dinv_i * (A @ ys) + 2*dinv_i^2*xw + b  (Ah=A+2I folded algebraically)
__global__ void k_gcn_epi(const float* __restrict__ agg,
                          const float* __restrict__ xw,
                          const float* __restrict__ dinv,
                          const float* __restrict__ b,
                          float* __restrict__ out, int n, int c, int relu) {
  int idx = blockIdx.x * blockDim.x + threadIdx.x;
  if (idx >= n * c) return;
  int i = idx / c, j = idx % c;
  float di = dinv[i];
  float v  = di * agg[idx] + 2.0f * di * di * xw[idx] + b[j];
  out[idx] = relu ? fmaxf(v, 0.0f) : v;
}

__global__ void k_pnorm(const float* __restrict__ p, float* __restrict__ pn,
                        int c) {
  if (blockIdx.x == 0 && threadIdx.x == 0) {
    float s = 0.0f;
    for (int j = 0; j < c; ++j) s += p[j] * p[j];
    pn[0] = sqrtf(s);
  }
}

__global__ void k_score(const float* __restrict__ h, const float* __restrict__ p,
                        const float* __restrict__ pn, float* __restrict__ sc,
                        int n, int c) {
  int i = blockIdx.x * blockDim.x + threadIdx.x;
  if (i >= n) return;
  float s = 0.0f;
  for (int j = 0; j < c; ++j) s += h[(size_t)i * c + j] * p[j];
  sc[i] = tanhf(s / pn[0]);
}

// exact top-k by rank counting; stable tie-break matches lax.top_k ordering
__global__ void k_rank_select(const float* __restrict__ sc,
                              int* __restrict__ perm, int n, int k) {
  int i = blockIdx.x * blockDim.x + threadIdx.x;
  if (i >= n) return;
  float si = sc[i];
  int r = 0;
  for (int j = 0; j < n; ++j) {
    float sj = sc[j];
    r += (sj > si) || (sj == si && j < i);
  }
  if (r < k) perm[r] = i;
}

__global__ void k_gather_x(const float* __restrict__ h,
                           const float* __restrict__ sc,
                           const int* __restrict__ perm,
                           float* __restrict__ xp, int k, int c) {
  int idx = blockIdx.x * blockDim.x + threadIdx.x;
  if (idx >= k * c) return;
  int r = idx / c, j = idx % c;
  int p = perm[r];
  xp[idx] = h[(size_t)p * c + j] * sc[p];
}

__global__ void k_gather_A(const half_t* __restrict__ A2,
                           const int* __restrict__ perm,
                           half_t* __restrict__ Ap, int k, int nsrc) {
  int idx = blockIdx.x * blockDim.x + threadIdx.x;
  if (idx >= k * k) return;
  int r = idx / k, c = idx % k;
  Ap[idx] = A2[(size_t)perm[r] * nsrc + perm[c]];
}

__global__ void k_diag_one(const half_t* __restrict__ A,
                           half_t* __restrict__ A1, int n) {
  int idx = blockIdx.x * blockDim.x + threadIdx.x;
  if (idx >= n * n) return;
  int i = idx / n, j = idx % n;
  A1[idx] = (i == j) ? (half_t)1.0f : A[idx];
}

__global__ void k_concat_res(const float* __restrict__ res,
                             float* __restrict__ cat, int n, int h) {
  int idx = blockIdx.x * blockDim.x + threadIdx.x;
  if (idx >= n * h) return;
  int i = idx / h, j = idx % h;
  cat[(size_t)i * 2 * h + j]     = res[idx];
  cat[(size_t)i * 2 * h + h + j] = 0.0f;
}

__global__ void k_scatter_up(const float* __restrict__ hup,
                             const int* __restrict__ perm,
                             float* __restrict__ cat, int k, int h) {
  int idx = blockIdx.x * blockDim.x + threadIdx.x;
  if (idx >= k * h) return;
  int r = idx / h, j = idx % h;
  cat[(size_t)perm[r] * 2 * h + h + j] = hup[idx];
}

// ---------------------------------------------------------------------------
// Host orchestration
// ---------------------------------------------------------------------------
static inline int gb(long n, int b) { return (int)((n + b - 1) / b); }

extern "C" void kernel_launch(void* const* d_in, const int* in_sizes, int n_in,
                              void* d_out, int out_size, void* d_ws, size_t ws_size,
                              hipStream_t stream) {
  (void)in_sizes; (void)n_in; (void)out_size; (void)ws_size;

  const float* x  = (const float*)d_in[0];
  const int*   ei = (const int*)d_in[1];
  const float* Wd[4] = {(const float*)d_in[2], (const float*)d_in[4],
                        (const float*)d_in[6], (const float*)d_in[8]};
  const float* bd[4] = {(const float*)d_in[3], (const float*)d_in[5],
                        (const float*)d_in[7], (const float*)d_in[9]};
  const float* pv[3] = {(const float*)d_in[10], (const float*)d_in[11],
                        (const float*)d_in[12]};
  const float* Wu[3] = {(const float*)d_in[13], (const float*)d_in[15],
                        (const float*)d_in[17]};
  const float* bu[3] = {(const float*)d_in[14], (const float*)d_in[16],
                        (const float*)d_in[18]};
  float* dout = (float*)d_out;

  // workspace carve-up
  char*  w = (char*)d_ws;
  auto carve = [&](size_t bytes) -> void* {
    void* p = (void*)w;
    w += (bytes + 255) & ~(size_t)255;
    return p;
  };
  const int lvN[4] = {4096, 2048, 1024, 512};
  half_t* Alv[4];
  Alv[0] = (half_t*)carve((size_t)4096 * 4096 * 2);
  Alv[1] = (half_t*)carve((size_t)2048 * 2048 * 2);
  Alv[2] = (half_t*)carve((size_t)1024 * 1024 * 2);
  Alv[3] = (half_t*)carve((size_t)512 * 512 * 2);
  half_t* S1    = (half_t*)carve((size_t)4096 * 4096 * 2);  // A1 (diag=1), symmetric
  half_t* S2    = (half_t*)carve((size_t)4096 * 4096 * 2);  // A2 = A1@A1
  half_t* hin16 = (half_t*)carve((size_t)4096 * 128 * 2);
  half_t* Wt16  = (half_t*)carve((size_t)128 * 64 * 2);     // W^T  (cout x cin)
  half_t* ysT   = (half_t*)carve((size_t)64 * 4096 * 2);    // ys^T (cout x n)
  float*  xw    = (float*)carve((size_t)4096 * 64 * 4);
  float*  agg   = (float*)carve((size_t)4096 * 64 * 4);
  float*  xs0   = (float*)carve((size_t)4096 * 64 * 4);
  float*  xs1   = (float*)carve((size_t)2048 * 64 * 4);
  float*  xs2   = (float*)carve((size_t)1024 * 64 * 4);
  float*  hP    = (float*)carve((size_t)4096 * 64 * 4);
  float*  hQ    = (float*)carve((size_t)4096 * 64 * 4);
  float*  cat   = (float*)carve((size_t)4096 * 128 * 4);
  float*  dinv  = (float*)carve((size_t)4096 * 4);
  float*  score = (float*)carve((size_t)4096 * 4);
  float*  pn    = (float*)carve(256);
  int* perm[3];
  perm[0] = (int*)carve((size_t)2048 * 4);
  perm[1] = (int*)carve((size_t)1024 * 4);
  perm[2] = (int*)carve((size_t)512 * 4);

  auto gemmF = [&](const half_t* A, const half_t* Bt, float* C,
                   int M, int Nn, int K) {
    int waves = (M / 32) * (Nn / 32);
    k_gemm_f16<0><<<gb(waves, 4), 128, 0, stream>>>(A, Bt, C, nullptr, M, Nn, K);
  };
  auto gemmH_diag0 = [&](const half_t* A, const half_t* Bt, half_t* C, int n) {
    int waves = (n / 32) * (n / 32);
    k_gemm_f16<1><<<gb(waves, 4), 128, 0, stream>>>(A, Bt, nullptr, C, n, n, n);
  };

  // GCN(improved=True): out = D^-1/2 (A+2I) D^-1/2 (hin @ W) + b
  auto gcn = [&](const half_t* A, int n, const float* hin, int cin,
                 const float* W, const float* b, int cout, int relu,
                 float* outp) {
    k_rowsum_dinv<<<gb(n, 256), 256, 0, stream>>>(A, dinv, n);
    k_f32_to_f16<<<gb((long)n * cin, 256), 256, 0, stream>>>(hin, hin16, n * cin);
    k_f32_to_f16_T<<<gb((long)cin * cout, 256), 256, 0, stream>>>(W, Wt16, cin, cout);
    gemmF(hin16, Wt16, xw, n, cout, cin);                // xw = hin @ W
    k_scale_rows_f16_T<<<gb((long)n * cout, 256), 256, 0, stream>>>(xw, dinv, ysT, n, cout);
    gemmF(A, ysT, agg, n, cout, n);                      // agg = A @ (dinv*xw)
    k_gcn_epi<<<gb((long)n * cout, 256), 256, 0, stream>>>(agg, xw, dinv, b,
                                                           outp, n, cout, relu);
  };

  // ---- build adjacency ----
  k_fill_h<<<gb((long)N0 * N0, 256), 256, 0, stream>>>(Alv[0], N0 * N0);
  k_build_adj<<<gb(NEDGE, 256), 256, 0, stream>>>(ei, Alv[0], NEDGE, N0);

  // ---- h0 = relu(gcn(x, A0)) ----
  gcn(Alv[0], N0, x, INC, Wd[0], bd[0], HIDC, 1, xs0);
  const float* curH = xs0;
  float* xsArr[3] = {xs0, xs1, xs2};

  // ---- down path ----
  for (int i = 0; i < 3; ++i) {
    int n = lvN[i], k = lvN[i + 1];
    // augment: A2 = (A+I)@(A+I), diag zeroed. A+I is symmetric -> its own
    // transpose serves as the column-major B operand.
    k_diag_one<<<gb((long)n * n, 256), 256, 0, stream>>>(Alv[i], S1, n);
    gemmH_diag0(S1, S1, S2, n);
    // TopKPooling
    k_pnorm<<<1, 64, 0, stream>>>(pv[i], pn, HIDC);
    k_score<<<gb(n, 256), 256, 0, stream>>>(curH, pv[i], pn, score, n, HIDC);
    k_rank_select<<<gb(n, 256), 256, 0, stream>>>(score, perm[i], n, k);
    k_gather_x<<<gb((long)k * HIDC, 256), 256, 0, stream>>>(curH, score, perm[i],
                                                            hP, k, HIDC);
    k_gather_A<<<gb((long)k * k, 256), 256, 0, stream>>>(S2, perm[i],
                                                         Alv[i + 1], k, n);
    float* dst = (i < 2) ? xsArr[i + 1] : hQ;
    gcn(Alv[i + 1], k, hP, HIDC, Wd[i + 1], bd[i + 1], HIDC, 1, dst);
    curH = dst;
  }

  // ---- up path (sum_res=False -> concat [res, scatter(h)]) ----
  for (int i = 0; i < 3; ++i) {
    int j = 2 - i;
    int n = lvN[j], k = lvN[j + 1];
    const float* res = xsArr[j];
    k_concat_res<<<gb((long)n * HIDC, 256), 256, 0, stream>>>(res, cat, n, HIDC);
    k_scatter_up<<<gb((long)k * HIDC, 256), 256, 0, stream>>>(curH, perm[j], cat,
                                                              k, HIDC);
    int cout  = (i < 2) ? HIDC : OUTC;
    int relu  = (i < 2) ? 1 : 0;
    float* dst = (i == 2) ? dout : ((i == 0) ? hP : hQ);
    gcn(Alv[j], n, cat, 2 * HIDC, Wu[i], bu[i], cout, relu, dst);
    curH = dst;
  }
}
